// NgramMultiheadAttention_50929722196231
// MI455X (gfx1250) — matrix operand order repaired and checked by reading, compiled, not run on hardware
//
#include <hip/hip_runtime.h>
#include <hip/hip_bf16.h>

// ---------------------------------------------------------------------------
// NgramMultiheadAttention for MI455X (gfx1250), wave32, WMMA bf16.
// T=512 B=4 E=1024 H=16 d=64 NGRAM=2; tgt_len=1536, rows=6144.
// ---------------------------------------------------------------------------

#define T_      512
#define B_      4
#define H_      16
#define D_      64
#define E_      1024
#define TGT_    1536        // (1+NGRAM)*T
#define ROWS_   6144        // TGT_*B_
#define BH_     64          // B_*H_
#define NBKT_   32
#define NEG_    (-10000.0f)

typedef __attribute__((ext_vector_type(16))) __bf16 v16bf;
typedef __attribute__((ext_vector_type(8)))  __bf16 bf16x8;
typedef __attribute__((ext_vector_type(8)))  float  v8f;
typedef __attribute__((ext_vector_type(4)))  float  f32x4;

union FragU { v16bf v; bf16x8 h[2]; };

// A-fragment (16x32 bf16, MxK): lane L: m = L&15, half = L>>4
//   element e: K = (e<8?0:16) + 8*half + (e&7)  -> two contiguous 8-elem chunks
static __device__ __forceinline__ v16bf load_a_frag(const __bf16* prow, int lane, int k0) {
    int half = lane >> 4;
    const __bf16* p = prow + k0 + half * 8;
    FragU u;
    u.h[0] = *(const bf16x8*)(p);
    u.h[1] = *(const bf16x8*)(p + 16);
    return u.v;
}

// B-fragment (32x16 bf16, KxN): lane L: n = L&15, half = L>>4
//   element e: K = (e<8?0:16) + 8*half + (e&7)  -> two contiguous 8-elem chunks
static __device__ __forceinline__ v16bf load_b_frag(const __bf16* prow, int lane, int k0) {
    int half = lane >> 4;
    const __bf16* p = prow + k0 + half * 8;
    FragU u;
    u.h[0] = *(const bf16x8*)(p);
    u.h[1] = *(const bf16x8*)(p + 16);
    return u.v;
}

static __device__ __forceinline__ v8f wmma_bf16(v16bf a, v16bf b, v8f c) {
    return __builtin_amdgcn_wmma_f32_16x16x32_bf16(false, a, false, b, (short)0, c, false, false);
}

// Uniform-base + unsigned-byte-offset loads -> saddr-form global loads.
static __device__ __forceinline__ float ldg_f32(const float* base, unsigned boff) {
    return *(const float*)((const char*)base + boff);
}
static __device__ __forceinline__ int ldg_i32(const int* base, unsigned boff) {
    return *(const int*)((const char*)base + boff);
}

// ---------------------------------------------------------------------------
__global__ void cast_f32_bf16(const float* __restrict__ x, __bf16* __restrict__ y, long n) {
    long i = (long)blockIdx.x * blockDim.x + threadIdx.x;
    if (i < n) y[i] = (__bf16)x[i];
}

// ---------------------------------------------------------------------------
// Generic GEMM:  C[M x N] = A_bf16 (strided rows) @ Wt_bf16(N x K, row-major)^T + bias
// Row r of A at:  A + (r%c0)*s0 + ((r/c0)%c1)*s1 + (r/(c0*c1))*s2   (elements)
// Block: 256 threads = 8 waves; wave w owns a 16 x 64 tile (4 WMMA tiles share one
// A-fragment per k-step: 4x A reuse). Block tile: 16 x 512.
template <typename OUT>
__global__ void gemm_bf16(const __bf16* __restrict__ A, const __bf16* __restrict__ Wt,
                          const float* __restrict__ bias, OUT* __restrict__ C,
                          int M, int N, int K,
                          int c0, long s0, int c1, long s1, long s2) {
    int lane = threadIdx.x & 31;
    int wave = threadIdx.x >> 5;
    int m0 = blockIdx.x * 16;
    int n0 = blockIdx.y * 512 + wave * 64;
    if (m0 >= M || n0 >= N) return;

    int r = m0 + (lane & 15);
    long arow = (long)(r % c0) * s0 + (long)((r / c0) % c1) * s1 + (long)(r / (c0 * c1)) * s2;
    const __bf16* Ap = A + arow;
    const __bf16* Bp = Wt + (long)(n0 + (lane & 15)) * K;   // tile j adds j*16*K

    v8f acc0 = {}, acc1 = {}, acc2 = {}, acc3 = {};
    for (int k0 = 0; k0 < K; k0 += 32) {
        if (k0 + 32 < K) __builtin_prefetch(Ap + k0 + 32, 0, 3);
        v16bf a  = load_a_frag(Ap, lane, k0);
        v16bf b0 = load_b_frag(Bp,                 lane, k0);
        v16bf b1 = load_b_frag(Bp + (long)16 * K,  lane, k0);
        v16bf b2 = load_b_frag(Bp + (long)32 * K,  lane, k0);
        v16bf b3 = load_b_frag(Bp + (long)48 * K,  lane, k0);
        acc0 = wmma_bf16(a, b0, acc0);
        acc1 = wmma_bf16(a, b1, acc1);
        acc2 = wmma_bf16(a, b2, acc2);
        acc3 = wmma_bf16(a, b3, acc3);
    }

    int half = lane >> 4;
    int nl = lane & 15;
#pragma unroll
    for (int j = 0; j < 4; ++j) {
        v8f acc = (j == 0) ? acc0 : (j == 1) ? acc1 : (j == 2) ? acc2 : acc3;
        int nn = n0 + j * 16 + nl;
        float bv = bias ? bias[nn] : 0.0f;
#pragma unroll
        for (int rr = 0; rr < 8; ++rr) {
            int mm = m0 + rr + 8 * half;
            C[(long)mm * N + nn] = (OUT)(acc[rr] + bv);
        }
    }
}

// ---------------------------------------------------------------------------
// Repack K rows and V^T per (b,h) from the qkv buffer.
__global__ void build_kv(const __bf16* __restrict__ qkv,
                         __bf16* __restrict__ Kc, __bf16* __restrict__ Vt) {
    long idx = (long)blockIdx.x * blockDim.x + threadIdx.x;
    if (idx >= (long)BH_ * TGT_ * D_) return;
    int j  = (int)(idx & (D_ - 1));
    long r = idx >> 6;
    int tg = (int)(r % TGT_);
    int bh = (int)(r / TGT_);
    int b = bh >> 4, h = bh & 15;
    const __bf16* src = qkv + ((long)tg * B_ + b) * (3 * E_) + h * D_ + j;
    Kc[((long)bh * TGT_ + tg) * D_ + j] = src[E_];
    Vt[((long)bh * D_ + j) * TGT_ + tg] = src[2 * E_];
}

// ---------------------------------------------------------------------------
// Fused attention: scores (WMMA) + rel gather + mask + softmax (LDS) + PV (WMMA).
// grid: x = T/16 t-tiles, y = BH, z = region (0 = main S=512; 1,2 = ngram S=1024)
// block: 256 threads (8 waves). LDS: 16 x 1024 fp32 score stripe = 64 KB.
// Causality: fully-masked tiles get NEG directly (expf underflows to exact 0),
// and the corresponding PV K-chunks are skipped -- bitwise-identical result.
// Gathers: uniform SGPR base + unsigned 32-bit byte offset (saddr-form loads),
// rr-dependent terms folded into instruction immediate offsets.
__global__ void attn_fused(const __bf16* __restrict__ qkv,
                           const __bf16* __restrict__ Kc, const __bf16* __restrict__ Vt,
                           const float* __restrict__ vals_main, const float* __restrict__ vals_ng,
                           const int* __restrict__ ib_main, const int* __restrict__ ib_ng,
                           const float* __restrict__ mask_main, const float* __restrict__ mask_ng,
                           __bf16* __restrict__ attnbf) {
    __shared__ float sc[16 * 1024];

    int tt = blockIdx.x, bh = blockIdx.y, reg = blockIdx.z;
    int b = bh >> 4, h = bh & 15;
    int t0 = tt * 16;
    int S = (reg == 0) ? T_ : 2 * T_;
    int ng = reg - 1;
    int tg0 = (reg == 0) ? t0 : (T_ + ng * T_ + t0);
    const float scale = 0.125f;   // d^-0.5, d=64

    int lane = threadIdx.x & 31;
    int wave = threadIdx.x >> 5;
    int half = lane >> 4;
    int nl = lane & 15;

    // Uniform gather base pointers (row length == S in both regions).
    const int*   ibp = ((reg == 0) ? ib_main : ib_ng) + (long)b * T_ * S;
    const float* mp  = (reg == 0) ? mask_main : (mask_ng + (long)ng * T_ * (2 * T_));
    const float* vph = ((reg == 0) ? (vals_main + (long)b * T_ * NBKT_ * H_)
                                   : (vals_ng + ((long)ng * B_ + b) * T_ * NBKT_ * H_)) + h;
    // Unsigned byte-offset bases; rr-dependent terms are constant adders.
    unsigned soff0 = (unsigned)((t0 + 8 * half) * S) * 4u;            // + rr*S*4 + sn*4
    unsigned voff0 = (unsigned)((t0 + 8 * half) * (NBKT_ * H_)) * 4u; // + rr*2048 + bkt*64

    // Q A-fragments for this 16-row tile (row = tg0 + (lane&15), stride B*3E)
    const __bf16* Qp = qkv + ((long)(tg0 + nl) * B_ + b) * (3 * E_) + h * D_;
    v16bf qa0 = load_a_frag(Qp, lane, 0);
    v16bf qa1 = load_a_frag(Qp, lane, 32);

    // ---- score phase: each wave strides over S/16 column tiles ----
    for (int st = wave; st < (S >> 4); st += 8) {
        int s0 = st * 16;
        int sn = s0 + nl;
        bool dead = (s0 < T_) ? (s0 > t0 + 15) : (s0 != T_ + t0);
        if (dead) {
#pragma unroll
            for (int rr = 0; rr < 8; ++rr) sc[(rr + 8 * half) * 1024 + sn] = NEG_;
            continue;
        }
        int s0p = s0 + ((reg == 2 && s0 >= T_) ? T_ : 0);   // physical row in Kc
        const __bf16* Kp = Kc + ((long)bh * TGT_ + s0p + nl) * D_;
        v16bf kb0 = load_b_frag(Kp, lane, 0);
        v16bf kb1 = load_b_frag(Kp, lane, 32);
        v8f acc = {};
        acc = wmma_bf16(qa0, kb0, acc);
        acc = wmma_bf16(qa1, kb1, acc);
        unsigned so = soff0 + (unsigned)sn * 4u;
#pragma unroll
        for (int rr = 0; rr < 8; ++rr) {
            unsigned rowb = so + (unsigned)(rr * S) * 4u;
            int bkt = ldg_i32(ibp, rowb);
            unsigned vb = voff0 + (unsigned)rr * (NBKT_ * H_ * 4u) + (unsigned)bkt * (H_ * 4u);
            float v = acc[rr] * scale + ldg_f32(vph, vb) + ldg_f32(mp, rowb);
            sc[(rr + 8 * half) * 1024 + sn] = v;
        }
    }
    __syncthreads();

    // ---- softmax: wave w owns rows 2w, 2w+1; wave32 shuffle reduction ----
#pragma unroll
    for (int rr2 = 0; rr2 < 2; ++rr2) {
        int m = wave * 2 + rr2;
        float mx = -1e30f;
        for (int s = lane; s < S; s += 32) mx = fmaxf(mx, sc[m * 1024 + s]);
        for (int off = 16; off; off >>= 1) mx = fmaxf(mx, __shfl_xor(mx, off, 32));
        float sum = 0.0f;
        for (int s = lane; s < S; s += 32) {
            float e = __expf(sc[m * 1024 + s] - mx);
            sc[m * 1024 + s] = e;
            sum += e;
        }
        for (int off = 16; off; off >>= 1) sum += __shfl_xor(sum, off, 32);
        float inv = 1.0f / sum;
        for (int s = lane; s < S; s += 32) sc[m * 1024 + s] *= inv;
    }
    __syncthreads();

    // ---- PV phase: waves 0..3 each own a 16-wide slice of head dim ----
    if (wave < 4) {
        const __bf16* Vp = Vt + ((long)bh * D_ + wave * 16 + nl) * TGT_;
        int kdiag = (T_ + t0) & ~31;          // 32-chunk holding the ngram diagonal
        v8f acc = {};
        for (int k0 = 0; k0 < S; k0 += 32) {
            bool dead = (k0 < T_) ? (k0 > t0 + 15) : (k0 != kdiag);
            if (dead) continue;               // prob == exact 0 for whole chunk
            // A-fragment from LDS probs: two contiguous 8-float chunks per lane.
            const float* rowp = sc + nl * 1024 + k0 + half * 8;
            f32x4 c0 = *(const f32x4*)(rowp);
            f32x4 c1 = *(const f32x4*)(rowp + 4);
            f32x4 c2 = *(const f32x4*)(rowp + 16);
            f32x4 c3 = *(const f32x4*)(rowp + 20);
            FragU ua;
#pragma unroll
            for (int e = 0; e < 4; ++e) {
                ua.v[e]      = (__bf16)c0[e];
                ua.v[e + 4]  = (__bf16)c1[e];
                ua.v[e + 8]  = (__bf16)c2[e];
                ua.v[e + 12] = (__bf16)c3[e];
            }
            int k0p = k0 + ((reg == 2 && k0 >= T_) ? T_ : 0);
            v16bf vb = load_b_frag(Vp, lane, k0p);
            acc = wmma_bf16(ua.v, vb, acc);
        }
        int j = wave * 16 + nl;
#pragma unroll
        for (int rr = 0; rr < 8; ++rr) {
            int tg = tg0 + rr + 8 * half;
            attnbf[((long)tg * B_ + b) * E_ + h * D_ + j] = (__bf16)acc[rr];
        }
    }
}

// ---------------------------------------------------------------------------
extern "C" void kernel_launch(void* const* d_in, const int* in_sizes, int n_in,
                              void* d_out, int out_size, void* d_ws, size_t ws_size,
                              hipStream_t stream) {
    const float* hidden   = (const float*)d_in[0];   // (1536, 4, 1024)
    const float* w_in     = (const float*)d_in[1];   // (3072, 1024)
    const float* b_in     = (const float*)d_in[2];   // (3072)
    const float* rel_w    = (const float*)d_in[3];   // (512, 1024)
    const float* rel_b    = (const float*)d_in[4];   // (512)
    const float* out_w    = (const float*)d_in[5];   // (1024, 1024)
    const float* out_b    = (const float*)d_in[6];   // (1024)
    const float* mask_m   = (const float*)d_in[7];   // (512, 512)
    const float* mask_ng  = (const float*)d_in[8];   // (2, 512, 1024)
    const int*   ib_main  = (const int*)d_in[9];     // (4, 512, 512)
    const int*   ib_ng    = (const int*)d_in[10];    // (4, 512, 1024)
    float* out = (float*)d_out;

    size_t off = 0;
    auto carve = [&](size_t bytes) -> char* {
        char* p = (char*)d_ws + off;
        off += (bytes + 255) & ~(size_t)255;
        return p;
    };
    __bf16* hbf     = (__bf16*)carve((size_t)ROWS_ * E_ * 2);
    __bf16* wqkv    = (__bf16*)carve((size_t)3 * E_ * E_ * 2);
    __bf16* wrel    = (__bf16*)carve((size_t)NBKT_ * H_ * E_ * 2);
    __bf16* wout    = (__bf16*)carve((size_t)E_ * E_ * 2);
    __bf16* qkv     = (__bf16*)carve((size_t)ROWS_ * 3 * E_ * 2);
    __bf16* Kc      = (__bf16*)carve((size_t)BH_ * TGT_ * D_ * 2);
    __bf16* Vt      = (__bf16*)carve((size_t)BH_ * D_ * TGT_ * 2);
    float*  valsM   = (float*)carve((size_t)B_ * T_ * NBKT_ * H_ * 4);
    float*  valsN   = (float*)carve((size_t)2 * B_ * T_ * NBKT_ * H_ * 4);
    __bf16* attnbf  = (__bf16*)carve((size_t)ROWS_ * E_ * 2);

    // --- 1. fp32 -> bf16 casts ---
    {
        long n = (long)ROWS_ * E_;
        cast_f32_bf16<<<dim3((n + 255) / 256), dim3(256), 0, stream>>>(hidden, hbf, n);
        n = (long)3 * E_ * E_;
        cast_f32_bf16<<<dim3((n + 255) / 256), dim3(256), 0, stream>>>(w_in, wqkv, n);
        n = (long)NBKT_ * H_ * E_;
        cast_f32_bf16<<<dim3((n + 255) / 256), dim3(256), 0, stream>>>(rel_w, wrel, n);
        n = (long)E_ * E_;
        cast_f32_bf16<<<dim3((n + 255) / 256), dim3(256), 0, stream>>>(out_w, wout, n);
    }

    // --- 2. QKV projection: (6144 x 1024) @ (3072 x 1024)^T -> bf16 qkv ---
    gemm_bf16<__bf16><<<dim3(ROWS_ / 16, (3 * E_) / 512), dim3(256), 0, stream>>>(
        hbf, wqkv, b_in, qkv, ROWS_, 3 * E_, E_,
        ROWS_, (long)E_, 1, 0L, 0L);

    // --- 3. rel values. main: rows (b*T+t) read hidden[(t*B+b)*E] ---
    gemm_bf16<float><<<dim3((B_ * T_) / 16, 1), dim3(256), 0, stream>>>(
        hbf, wrel, rel_b, valsM, B_ * T_, NBKT_ * H_, E_,
        T_, (long)B_ * E_, B_, (long)E_, 0L);
    //     ngram: rows ((ng*B+b)*T+t) read hidden[(T + ng*T + t)*B*E + b*E]
    gemm_bf16<float><<<dim3((2 * B_ * T_) / 16, 1), dim3(256), 0, stream>>>(
        hbf + (long)T_ * B_ * E_, wrel, rel_b, valsN, 2 * B_ * T_, NBKT_ * H_, E_,
        T_, (long)B_ * E_, B_, (long)E_, (long)T_ * B_ * E_);

    // --- 4. K / V^T repack per (b,h) ---
    {
        long n = (long)BH_ * TGT_ * D_;
        build_kv<<<dim3((n + 255) / 256), dim3(256), 0, stream>>>(qkv, Kc, Vt);
    }

    // --- 5. fused attention (scores + rel + mask + softmax + PV) ---
    attn_fused<<<dim3(T_ / 16, BH_, 3), dim3(256), 0, stream>>>(
        qkv, Kc, Vt, valsM, valsN, ib_main, ib_ng, mask_m, mask_ng, attnbf);

    // --- 6. output projection straight into d_out (row order matches concat) ---
    gemm_bf16<float><<<dim3(ROWS_ / 16, E_ / 512), dim3(256), 0, stream>>>(
        attnbf, wout, out_b, out, ROWS_, E_, E_,
        ROWS_, (long)E_, 1, 0L, 0L);

    (void)in_sizes; (void)n_in; (void)out_size; (void)ws_size;
}
